// Attention_24687472017666
// MI455X (gfx1250) — matrix operand order repaired and checked by reading, not compiled, not run on hardware
//
#include <hip/hip_runtime.h>
#include <cmath>

typedef __attribute__((ext_vector_type(16))) _Float16 v16h;
typedef __attribute__((ext_vector_type(8)))  _Float16 v8h;
typedef __attribute__((ext_vector_type(4)))  _Float16 v4h;
typedef __attribute__((ext_vector_type(8)))  float    v8f;
typedef _Float16 h16;

#define WMMA_F16(a,b,c) __builtin_amdgcn_wmma_f32_16x16x32_f16(false,(a),false,(b),(short)0,(c),false,false)

static constexpr int BB  = 2;
static constexpr int T   = 2048;
static constexpr int CD  = 1024;   // channel dim C
static constexpr int H   = 16;
static constexpr int DH  = 64;
static constexpr int MR  = BB * T;     // 4096 rows (b,t)
static constexpr int NK  = H * DH;     // 1024 (h,d)
static constexpr float SOFT_CAP = 50.0f;

// ---- fragment load helpers: provably aligned vector loads --------------------
__device__ __forceinline__ v16h frag_cat(v8h lo, v8h hi)
{
    return __builtin_shufflevector(lo, hi, 0,1,2,3,4,5,6,7,8,9,10,11,12,13,14,15);
}
__device__ __forceinline__ v16h frag_load32(const h16* p)     // 32B contiguous
{
    return *(const v16h*)p;
}
__device__ __forceinline__ v16h frag_load2(const h16* p0, const h16* p1)
{
    return frag_cat(*(const v8h*)p0, *(const v8h*)p1);        // 2 x 16B
}

// branch-free soft-cap: 50*tanh(x/50) = 50*(e^z-1)*rcp(e^z+1), z = 2x/50 clamped
__device__ __forceinline__ float softcap(float x)
{
    float z = fminf(fmaxf(x * (2.0f / SOFT_CAP), -30.0f), 30.0f);
    float e = __expf(z);
    return SOFT_CAP * (e - 1.0f) * __builtin_amdgcn_rcpf(e + 1.0f);
}

// ---------------------------------------------------------------------------
// f32 -> f16 straight conversion (x), 4 elements per thread
// ---------------------------------------------------------------------------
__global__ __launch_bounds__(256)
void cvt_f16_kernel(const float* __restrict__ src, h16* __restrict__ dst)
{
    const int i = (blockIdx.x * blockDim.x + threadIdx.x) * 4;
    const float4 v = *(const float4*)(src + i);
    v4h o;
    o[0] = (h16)v.x; o[1] = (h16)v.y; o[2] = (h16)v.z; o[3] = (h16)v.w;
    *(v4h*)(dst + i) = o;
}

// ---------------------------------------------------------------------------
// Transpose f32 [K,N] -> f16 [N,K] through LDS (coalesced both sides)
// ---------------------------------------------------------------------------
__global__ __launch_bounds__(256)
void transpose_f16_kernel(const float* __restrict__ src, h16* __restrict__ dst,
                          int K, int N)
{
    __shared__ h16 tile[32][33];
    const int k0 = blockIdx.y * 32;
    const int n0 = blockIdx.x * 32;
    const int tx = threadIdx.x;            // 0..31
#pragma unroll
    for (int r = threadIdx.y; r < 32; r += 8)
        tile[r][tx] = (h16)src[(size_t)(k0 + r) * N + n0 + tx];
    __syncthreads();
#pragma unroll
    for (int r = threadIdx.y; r < 32; r += 8)
        dst[(size_t)(n0 + r) * K + k0 + tx] = tile[tx][r];
}

// ---------------------------------------------------------------------------
// GEMM: C[M,N] = A[M,K] * Bt^T, A f16 row-major [M,K], Bt f16 N-major [N,K].
// f32 accumulate via WMMA; one wave -> 16x64 strip (A fragment reused 4x).
// Fragment layouts (C/D per CDNA5 ISA 7.12.2; A/B use this code's contiguous k order):
//   A 16x32 f16 : lane = row;  half i -> K = i + 16*grp (same order as B)
//   B 32x16 f16 : lane = col;  half h -> K = h + 16*grp (this code's order, not the ISA table)
//   C/D 16x16   : vgpr g, lane l -> row g + 8*grp, col lane&15
// STORE_MODE: 0 = f16 row-major, 1 = f16 per-head-transposed (for V), 2 = f32
// ---------------------------------------------------------------------------
template<int STORE_MODE>
__global__ __launch_bounds__(256)
void gemm_wmma_kernel(const h16* __restrict__ A, const h16* __restrict__ Bt,
                      void* __restrict__ Co, int M, int N, int K)
{
    const int lane = threadIdx.x;
    const int ln   = lane & 15;
    const int grp  = lane >> 4;
    const int tile = blockIdx.x * blockDim.y + threadIdx.y;
    const int ntn  = N >> 6;                  // 64-wide strips along N
    const int m0   = (tile / ntn) << 4;
    const int n0   = (tile % ntn) << 6;
    if (m0 >= M) return;

    const h16* Arow = A  + (size_t)(m0 + ln) * K;
    const h16* Brow = Bt + (size_t)(n0 + ln) * K;   // + nt*16*K per sub-tile

    v8f acc0 = {}, acc1 = {}, acc2 = {}, acc3 = {};
    for (int k0 = 0; k0 < K; k0 += 32) {
        const v16h a  = frag_load32(Arow + k0 + 16 * grp);
        const h16* pb = Brow + k0 + 16 * grp;
        const v16h b0 = frag_load32(pb);
        const v16h b1 = frag_load32(pb + (size_t)16 * K);
        const v16h b2 = frag_load32(pb + (size_t)32 * K);
        const v16h b3 = frag_load32(pb + (size_t)48 * K);
        acc0 = WMMA_F16(a, b0, acc0);
        acc1 = WMMA_F16(a, b1, acc1);
        acc2 = WMMA_F16(a, b2, acc2);
        acc3 = WMMA_F16(a, b3, acc3);
    }

#pragma unroll
    for (int g = 0; g < 8; ++g) {
        const int row = m0 + g + 8 * grp;
        const float r0 = acc0[g], r1 = acc1[g], r2 = acc2[g], r3 = acc3[g];
        if (STORE_MODE == 0) {
            h16* C = (h16*)Co;
            C[(size_t)row * N + n0 +  0 + ln] = (h16)r0;
            C[(size_t)row * N + n0 + 16 + ln] = (h16)r1;
            C[(size_t)row * N + n0 + 32 + ln] = (h16)r2;
            C[(size_t)row * N + n0 + 48 + ln] = (h16)r3;
        } else if (STORE_MODE == 2) {
            float* C = (float*)Co;
            C[(size_t)row * N + n0 +  0 + ln] = r0;
            C[(size_t)row * N + n0 + 16 + ln] = r1;
            C[(size_t)row * N + n0 + 32 + ln] = r2;
            C[(size_t)row * N + n0 + 48 + ln] = r3;
        } else {
            // V transposed per head: vt[((b*H+h)*DH+d)*T + t]
            h16* C = (h16*)Co;
            const int b = row >> 11, t = row & (T - 1);
#pragma unroll
            for (int nt = 0; nt < 4; ++nt) {
                const int n = n0 + nt * 16 + ln;
                const int hh = n >> 6, d = n & (DH - 1);
                const float r = nt == 0 ? r0 : nt == 1 ? r1 : nt == 2 ? r2 : r3;
                C[(((size_t)b * H + hh) * DH + d) * T + t] = (h16)r;
            }
        }
    }
}

// ---------------------------------------------------------------------------
// RoPE on q (with 1/sqrt(D) scale) and k, in-place on f16 [m][h*64+d].
// ---------------------------------------------------------------------------
__global__ __launch_bounds__(256)
void rope_kernel(h16* __restrict__ q, h16* __restrict__ k)
{
    const int idx = blockIdx.x * blockDim.x + threadIdx.x;   // 2^21 total
    const int j  = idx & 31;
    const int hh = (idx >> 5) & (H - 1);
    const int t  = (idx >> 9) & (T - 1);
    const int b  = idx >> 20;
    const size_t off = (size_t)(b * T + t) * NK + hh * DH;

    const float inv_ts = powf(10000.0f, -(float)j * (1.0f / 32.0f));
    const float ang = (float)t * inv_ts;
    const float sn = sinf(ang), cs = cosf(ang);
    const float scale = 0.125f;   // 1/sqrt(64)

    float qf = (float)q[off + j], qs = (float)q[off + j + 32];
    q[off + j]      = (h16)((qf * cs - qs * sn) * scale);
    q[off + j + 32] = (h16)((qs * cs + qf * sn) * scale);

    float kf = (float)k[off + j], ks = (float)k[off + j + 32];
    k[off + j]      = (h16)(kf * cs - ks * sn);
    k[off + j + 32] = (h16)(ks * cs + kf * sn);
}

// ---------------------------------------------------------------------------
// Flash attention: one wave per 16 query rows of one (b,h).
// q,k: f16 [b*T+t][h*64+d];  v: f16 transposed [b][h][d][t];  o: like q.
// All fragment loads are batched ahead of the WMMA chains so they form one
// clause and overlap WMMA issue with outstanding loads.
// ---------------------------------------------------------------------------
__global__ __launch_bounds__(256)
void flash_kernel(const h16* __restrict__ q, const h16* __restrict__ k,
                  const h16* __restrict__ vt, h16* __restrict__ o)
{
    __shared__ h16 plds[8][16 * 32];   // per-wave P tile (16x32 f16)

    const int lane = threadIdx.x;
    const int ln   = lane & 15;
    const int grp  = lane >> 4;
    const int w    = threadIdx.y;
    const int gw   = blockIdx.x * 8 + w;
    const int qt = gw & 127;           // T/16 q-tiles
    const int bh = gw >> 7;
    const int hh = bh & 15;
    const int b  = bh >> 4;
    const int q0 = qt << 4;

    const size_t basebh = (size_t)b * T * NK + (size_t)hh * DH;
    const h16* vhead = vt + ((size_t)b * H + hh) * DH * T;   // [d][t]

    // Q A-fragments for d = 0..31 and 32..63 (loaded once)
    const h16* qrow = q + basebh + (size_t)(q0 + ln) * NK;
    const v16h qa0 = frag_load32(qrow + 16 * grp);
    const v16h qa1 = frag_load32(qrow + 32 + 16 * grp);

    v8f o0 = {}, o1 = {}, o2 = {}, o3 = {};
    float mrun[8], lrun[8];
#pragma unroll
    for (int g = 0; g < 8; ++g) { mrun[g] = -1e30f; lrun[g] = 0.0f; }

    const int kend = q0 + 16;               // causal: keys 0 .. q0+15
    for (int kb = 0; kb < kend; kb += 32) {
        // branchless prefetch of the (clamped) next K/V block
        const int kbn = min(kb + 32, kend - 16);
        __builtin_prefetch(k + basebh + (size_t)(kbn + ln) * NK, 0, 0);
        __builtin_prefetch(vhead + (size_t)ln * T + kbn, 0, 0);

        // ---- batch all four K fragments, then the four S WMMAs
        const h16* krow  = k + basebh + (size_t)(kb + ln) * NK;
        const h16* krow2 = krow + (size_t)16 * NK;
        const v16h k00 = frag_load32(krow  + 16 * grp);
        const v16h k01 = frag_load32(krow  + 32 + 16 * grp);
        const v16h k10 = frag_load32(krow2 + 16 * grp);
        const v16h k11 = frag_load32(krow2 + 32 + 16 * grp);
        v8f s0 = {}, s1 = {};
        s0 = WMMA_F16(qa0, k00, s0);
        s0 = WMMA_F16(qa1, k01, s0);
        s1 = WMMA_F16(qa0, k10, s1);
        s1 = WMMA_F16(qa1, k11, s1);

        // ---- soft-cap, causal mask, online softmax update
        float p0a[8], p1a[8];
#pragma unroll
        for (int g = 0; g < 8; ++g) {
            const int row = q0 + g + 8 * grp;
            float a0 = softcap(s0[g]);
            float a1 = softcap(s1[g]);
            if (kb + ln      > row) a0 = -1e30f;
            if (kb + 16 + ln > row) a1 = -1e30f;
            float rm = fmaxf(a0, a1);
#pragma unroll
            for (int d = 1; d < 16; d <<= 1) rm = fmaxf(rm, __shfl_xor(rm, d, 32));
            const float mnew  = fmaxf(mrun[g], rm);
            const float alpha = __expf(mrun[g] - mnew);
            mrun[g] = mnew;
            const float e0 = __expf(a0 - mnew);
            const float e1 = __expf(a1 - mnew);
            p0a[g] = e0; p1a[g] = e1;
            float rs = e0 + e1;
#pragma unroll
            for (int d = 1; d < 16; d <<= 1) rs += __shfl_xor(rs, d, 32);
            lrun[g] = lrun[g] * alpha + rs;
            o0[g] *= alpha; o1[g] *= alpha; o2[g] *= alpha; o3[g] *= alpha;
        }

        // ---- redistribute P: C-layout -> A-layout through LDS
        h16* pbuf = &plds[w][0];
#pragma unroll
        for (int g = 0; g < 8; ++g) {
            const int row = g + 8 * grp;
            pbuf[row * 32 + ln]      = (h16)p0a[g];
            pbuf[row * 32 + 16 + ln] = (h16)p1a[g];
        }
        asm volatile("s_wait_dscnt 0" ::: "memory");
        const h16* pr = pbuf + ln * 32;
        const v16h pa = frag_load2(pr + 16 * grp, pr + 16 * grp + 8);

        // ---- batch all four V fragments, then the four PV WMMAs
        const h16* vp = vhead + (size_t)ln * T + kb + 16 * grp;
        const v16h v0 = frag_load32(vp);
        const v16h v1 = frag_load32(vp + (size_t)16 * T);
        const v16h v2 = frag_load32(vp + (size_t)32 * T);
        const v16h v3 = frag_load32(vp + (size_t)48 * T);
        o0 = WMMA_F16(pa, v0, o0);
        o1 = WMMA_F16(pa, v1, o1);
        o2 = WMMA_F16(pa, v2, o2);
        o3 = WMMA_F16(pa, v3, o3);
    }

    // ---- normalize and store 16x64 output tile (f16 workspace)
#pragma unroll
    for (int g = 0; g < 8; ++g) {
        const int row = q0 + g + 8 * grp;
        const float invl = __builtin_amdgcn_rcpf(lrun[g]);
        h16* orow = o + basebh + (size_t)row * NK;
        orow[0  + ln] = (h16)(o0[g] * invl);
        orow[16 + ln] = (h16)(o1[g] * invl);
        orow[32 + ln] = (h16)(o2[g] * invl);
        orow[48 + ln] = (h16)(o3[g] * invl);
    }
}

// ---------------------------------------------------------------------------
extern "C" void kernel_launch(void* const* d_in, const int* in_sizes, int n_in,
                              void* d_out, int out_size, void* d_ws, size_t ws_size,
                              hipStream_t stream)
{
    const float* x  = (const float*)d_in[0];
    // d_in[1] = causal mask, handled analytically
    const float* wq = (const float*)d_in[2];
    const float* wk = (const float*)d_in[3];
    const float* wv = (const float*)d_in[4];
    const float* wo = (const float*)d_in[5];
    float* out = (float*)d_out;

    const size_t plane = (size_t)MR * NK;      // 4M halves
    const size_t wsz   = (size_t)CD * NK;      // 1M halves
    h16* xh  = (h16*)d_ws;
    h16* wqt = xh  + plane;
    h16* wkt = wqt + wsz;
    h16* wvt = wkt + wsz;
    h16* wot = wvt + wsz;
    h16* qws = wot + wsz;
    h16* kws = qws + plane;
    h16* vtw = kws + plane;
    h16* ows = vtw + plane;

    dim3 blk(32, 8);

    // 0) convert x to f16; transpose weights to N-major f16
    cvt_f16_kernel<<<(int)(plane / 4 / 256), 256, 0, stream>>>(x, xh);
    dim3 tgrid(CD / 32, NK / 32);
    transpose_f16_kernel<<<tgrid, blk, 0, stream>>>(wq, wqt, CD, NK);
    transpose_f16_kernel<<<tgrid, blk, 0, stream>>>(wk, wkt, CD, NK);
    transpose_f16_kernel<<<tgrid, blk, 0, stream>>>(wv, wvt, CD, NK);
    transpose_f16_kernel<<<tgrid, blk, 0, stream>>>(wo, wot, NK, CD);

    // 1) projections: each wave computes 16x64
    const int tiles = (MR / 16) * (NK / 64);   // 4096
    dim3 ggrid(tiles / 8);
    gemm_wmma_kernel<0><<<ggrid, blk, 0, stream>>>(xh, wqt, qws, MR, NK, CD);
    gemm_wmma_kernel<0><<<ggrid, blk, 0, stream>>>(xh, wkt, kws, MR, NK, CD);
    gemm_wmma_kernel<1><<<ggrid, blk, 0, stream>>>(xh, wvt, vtw, MR, NK, CD);

    // 2) RoPE (+ q scale)
    const int rope_threads = BB * T * H * 32;  // 2^21
    rope_kernel<<<rope_threads / 256, 256, 0, stream>>>(qws, kws);

    // 3) flash attention
    const int fwaves = BB * H * (T / 16);      // 4096
    flash_kernel<<<fwaves / 8, blk, 0, stream>>>(qws, kws, vtw, ows);

    // 4) output projection -> f32 d_out
    gemm_wmma_kernel<2><<<ggrid, blk, 0, stream>>>(ows, wot, out, MR, CD, NK);
}
